// GINGNN_76184129896629
// MI455X (gfx1250) — matrix operand order repaired and checked by reading, not compiled, not run on hardware
//
#include <hip/hip_runtime.h>
#include <stdint.h>

typedef __attribute__((ext_vector_type(16))) __bf16 v16bf;
typedef __attribute__((ext_vector_type(8)))  __bf16 v8bf;
typedef __attribute__((ext_vector_type(8)))  float  v8f;
typedef __attribute__((ext_vector_type(4)))  float  v4f;

#define NNODES 8192
#define INDIM  128
#define HDIM   256

// Native f32 -> bf16 (RNE) — lets the backend emit packed v_cvt ops instead
// of a 3-VALU bit-twiddle per element.
static __device__ __forceinline__ __bf16 f2bf(float f) {
  return (__bf16)f;
}

static __device__ __forceinline__ v16bf load_b16(const __bf16* __restrict__ p) {
  v8bf lo = *(const v8bf*)p;
  v8bf hi = *(const v8bf*)(p + 16);
  v16bf r;
#pragma unroll
  for (int i = 0; i < 8; ++i) { r[i] = lo[i]; r[i + 8] = hi[i]; }
  return r;
}

// -------- Fused (X [+ X2]) @ Wt^T + bias [,relu] via v_wmma_f32_16x16x32_bf16
// X: [N x K] fp32, Wt: [HDIM x K] bf16 (pre-transposed weight), Y: [N x HDIM] fp32
// grid.x = N/16, block = 256 (8 waves); wave w owns output columns [32w, 32w+32).
template <int K, bool ADD, bool RELU>
__global__ __launch_bounds__(256) void gemm_wmma_kernel(
    const float* __restrict__ X, const float* __restrict__ X2,
    const __bf16* __restrict__ Wt, const float* __restrict__ bias,
    float* __restrict__ Y)
{
  const int lane = threadIdx.x & 31;
  const int wv   = threadIdx.x >> 5;   // 0..7
  const int m    = lane & 15;
  const int half = lane >> 4;          // 0 or 1
  const int rowBase = blockIdx.x * 16;
  const size_t rowOff = (size_t)(rowBase + m) * K;
  const int col0 = wv * 32 + m;
  const int col1 = col0 + 16;
  const __bf16* wp0 = Wt + (size_t)col0 * K;
  const __bf16* wp1 = Wt + (size_t)col1 * K;

  v8f c0 = {};
  v8f c1 = {};

#pragma unroll 4
  for (int k0 = 0; k0 < K; k0 += 32) {
    // A fragment: lanes 0-15 -> K = k0+[0..7, 16..23]; lanes 16-31 -> +8
    const int ka = k0 + half * 8;
    v4f f0 = *(const v4f*)(X + rowOff + ka);
    v4f f1 = *(const v4f*)(X + rowOff + ka + 4);
    v4f f2 = *(const v4f*)(X + rowOff + ka + 16);
    v4f f3 = *(const v4f*)(X + rowOff + ka + 20);
    if constexpr (ADD) {
      f0 += *(const v4f*)(X2 + rowOff + ka);
      f1 += *(const v4f*)(X2 + rowOff + ka + 4);
      f2 += *(const v4f*)(X2 + rowOff + ka + 16);
      f3 += *(const v4f*)(X2 + rowOff + ka + 20);
    }
    v16bf a;
#pragma unroll
    for (int i = 0; i < 4; ++i) {
      a[i]      = f2bf(f0[i]);
      a[4 + i]  = f2bf(f1[i]);
      a[8 + i]  = f2bf(f2[i]);
      a[12 + i] = f2bf(f3[i]);
    }
    // B fragments: lanes 0-15 -> K = k0+[0..7, 16..23]; lanes 16-31 -> +8
    const int kb = k0 + half * 8;
    v16bf b0 = load_b16(wp0 + kb);
    v16bf b1 = load_b16(wp1 + kb);
    c0 = __builtin_amdgcn_wmma_f32_16x16x32_bf16(false, a, false, b0,
                                                 (short)0, c0, false, false);
    c1 = __builtin_amdgcn_wmma_f32_16x16x32_bf16(false, a, false, b1,
                                                 (short)0, c1, false, false);
  }

  const float bv0 = bias[col0];
  const float bv1 = bias[col1];
  const int mrow = rowBase + half * 8;   // C layout: M = v + 8*(lane>=16)
#pragma unroll
  for (int v = 0; v < 8; ++v) {
    float y0 = c0[v] + bv0;
    float y1 = c1[v] + bv1;
    if constexpr (RELU) { y0 = fmaxf(y0, 0.0f); y1 = fmaxf(y1, 0.0f); }
    Y[(size_t)(mrow + v) * HDIM + col0] = y0;
    Y[(size_t)(mrow + v) * HDIM + col1] = y1;
  }
}

// -------- segment_sum: one block per edge; src/dst become uniform scalar loads.
template <int HD>
__global__ __launch_bounds__(HD) void scatter_add_kernel(
    const float* __restrict__ X, const int* __restrict__ src,
    const int* __restrict__ dst, float* __restrict__ agg)
{
  const int e = blockIdx.x;
  const int f = threadIdx.x;
  const int s = src[e];
  const int d = dst[e];
  atomicAdd(agg + (size_t)d * HD + f, X[(size_t)s * HD + f]);
}

__global__ __launch_bounds__(256) void zero_kernel(float* __restrict__ p)
{
  size_t i = ((size_t)blockIdx.x * 256 + threadIdx.x) * 4;
  *(v4f*)(p + i) = (v4f){0.f, 0.f, 0.f, 0.f};
}

// Transpose + fp32->bf16 convert: W [K x 256] -> Wt [256 x K]
__global__ __launch_bounds__(256) void wt_convert_kernel(
    const float* __restrict__ W, __bf16* __restrict__ Wt, int K)
{
  int idx = blockIdx.x * 256 + threadIdx.x;   // over K*256
  int k = idx >> 8;
  int h = idx & 255;
  Wt[(size_t)h * K + k] = f2bf(W[idx]);
}

// li = X @ w_edge[:H], lj = X @ w_edge[H:]; one wave per row.
__global__ __launch_bounds__(256) void rowdot_kernel(
    const float* __restrict__ X, const float* __restrict__ we,
    float* __restrict__ li, float* __restrict__ lj)
{
  const int lane = threadIdx.x & 31;
  const int wv   = threadIdx.x >> 5;
  const int row  = blockIdx.x * 8 + wv;
  const float* xr = X + (size_t)row * HDIM;
  float s0 = 0.f, s1 = 0.f;
#pragma unroll
  for (int i = lane; i < HDIM; i += 32) {
    float v = xr[i];
    s0 += v * we[i];
    s1 += v * we[HDIM + i];
  }
#pragma unroll
  for (int off = 16; off > 0; off >>= 1) {
    s0 += __shfl_xor(s0, off, 32);
    s1 += __shfl_xor(s1, off, 32);
  }
  if (lane == 0) { li[row] = s0; lj[row] = s1; }
}

// out[i, j] = li[i] + lj[j] + b ; 256 MB streamed with non-temporal b128 stores.
__global__ __launch_bounds__(256) void outer_kernel(
    const float* __restrict__ li, const float* __restrict__ lj,
    const float* __restrict__ be, float* __restrict__ out)
{
  const int i = blockIdx.y;
  const int j = (blockIdx.x * 256 + threadIdx.x) * 4;
  const float a = li[i] + be[0];
  v4f l = *(const v4f*)(lj + j);
  v4f o = (v4f){a + l[0], a + l[1], a + l[2], a + l[3]};
  __builtin_nontemporal_store(o, (v4f*)(out + (size_t)i * NNODES + j));
}

extern "C" void kernel_launch(void* const* d_in, const int* in_sizes, int n_in,
                              void* d_out, int out_size, void* d_ws, size_t ws_size,
                              hipStream_t stream) {
  const float* x     = (const float*)d_in[0];
  const int*   edge  = (const int*)d_in[1];
  const float* w0a   = (const float*)d_in[2];
  const float* b0a   = (const float*)d_in[3];
  const float* w0b   = (const float*)d_in[4];
  const float* b0b   = (const float*)d_in[5];
  const float* w1a   = (const float*)d_in[6];
  const float* b1a   = (const float*)d_in[7];
  const float* w1b   = (const float*)d_in[8];
  const float* b1b   = (const float*)d_in[9];
  const float* wout  = (const float*)d_in[10];
  const float* bout  = (const float*)d_in[11];
  const float* wedge = (const float*)d_in[12];
  const float* bedge = (const float*)d_in[13];

  const int E = in_sizes[1] / 2;
  const int* src = edge;
  const int* dst = edge + E;

  char* ws = (char*)d_ws;
  size_t off = 0;
  auto alloc = [&](size_t bytes) -> char* {
    char* p = ws + off;
    off += (bytes + 255) & ~(size_t)255;
    return p;
  };
  float*  agg    = (float*)alloc((size_t)NNODES * HDIM * 4);
  float*  tmpB   = (float*)alloc((size_t)NNODES * HDIM * 4);
  float*  tmpC   = (float*)alloc((size_t)NNODES * HDIM * 4);
  float*  tmpD   = (float*)alloc((size_t)NNODES * HDIM * 4);
  float*  li     = (float*)alloc((size_t)NNODES * 4);
  float*  lj     = (float*)alloc((size_t)NNODES * 4);
  __bf16* w0a_t  = (__bf16*)alloc((size_t)INDIM * HDIM * 2);
  __bf16* w0b_t  = (__bf16*)alloc((size_t)HDIM * HDIM * 2);
  __bf16* w1a_t  = (__bf16*)alloc((size_t)HDIM * HDIM * 2);
  __bf16* w1b_t  = (__bf16*)alloc((size_t)HDIM * HDIM * 2);
  __bf16* wout_t = (__bf16*)alloc((size_t)HDIM * HDIM * 2);

  // Pre-transpose + convert weights to bf16 (stay hot in L2 across GEMMs).
  wt_convert_kernel<<<INDIM, 256, 0, stream>>>(w0a, w0a_t, INDIM);
  wt_convert_kernel<<<HDIM,  256, 0, stream>>>(w0b, w0b_t, HDIM);
  wt_convert_kernel<<<HDIM,  256, 0, stream>>>(w1a, w1a_t, HDIM);
  wt_convert_kernel<<<HDIM,  256, 0, stream>>>(w1b, w1b_t, HDIM);
  wt_convert_kernel<<<HDIM,  256, 0, stream>>>(wout, wout_t, HDIM);

  const int gemmGrid = NNODES / 16;

  // ---- GIN layer 0 (K = 128)
  zero_kernel<<<(NNODES * INDIM) / 1024, 256, 0, stream>>>(agg);
  scatter_add_kernel<INDIM><<<E, INDIM, 0, stream>>>(x, src, dst, agg);
  gemm_wmma_kernel<INDIM, true,  true><<<gemmGrid, 256, 0, stream>>>(x,    agg,     w0a_t, b0a, tmpB);
  gemm_wmma_kernel<HDIM,  false, true><<<gemmGrid, 256, 0, stream>>>(tmpB, nullptr, w0b_t, b0b, tmpC);

  // ---- GIN layer 1 (K = 256)
  zero_kernel<<<(NNODES * HDIM) / 1024, 256, 0, stream>>>(agg);
  scatter_add_kernel<HDIM><<<E, HDIM, 0, stream>>>(tmpC, src, dst, agg);
  gemm_wmma_kernel<HDIM, true,  true><<<gemmGrid, 256, 0, stream>>>(tmpC, agg,     w1a_t, b1a, tmpB);
  gemm_wmma_kernel<HDIM, false, true><<<gemmGrid, 256, 0, stream>>>(tmpB, nullptr, w1b_t, b1b, tmpD);

  // ---- output projection (no relu)
  gemm_wmma_kernel<HDIM, false, false><<<gemmGrid, 256, 0, stream>>>(tmpD, nullptr, wout_t, bout, tmpB);

  // ---- edge logits
  rowdot_kernel<<<NNODES / 8, 256, 0, stream>>>(tmpB, wedge, li, lj);
  dim3 og(NNODES / 1024, NNODES);
  outer_kernel<<<og, 256, 0, stream>>>(li, lj, bedge, (float*)d_out);
}
